// TemporalLSTM_67697274519660
// MI455X (gfx1250) — compile-verified, matrix operand order checked by reading, not run on hardware
//
#include <hip/hip_runtime.h>
#include <hip/hip_bf16.h>

typedef __bf16 v16bf __attribute__((ext_vector_type(16)));
typedef float  v8f   __attribute__((ext_vector_type(8)));

// Problem constants
#define BB   128
#define SS   512
#define DD   256
#define HSZ  512
#define TNN  128
#define ANN  128
#define TT   24
#define G4   2048      // 4*HS
#define KWP  288       // D+1=257 padded to 9*32 for WMMA K

// ---------------------------------------------------------------------------
// Branch-free tanh: 1 - 2/(exp(2x)+1). exp->inf gives +1, exp->0 gives -1.
// ---------------------------------------------------------------------------
__device__ __forceinline__ float fast_tanh(float x) {
    float e = __expf(2.0f * x);
    return 1.0f - 2.0f / (e + 1.0f);
}
__device__ __forceinline__ float fast_sigmoid(float x) {
    return 1.0f / (1.0f + __expf(-x));
}

// ---------------------------------------------------------------------------
// bf16 fragment loader: per-lane 16 bf16 elements as two contiguous 16B chunks
// (K layout: lanes 0-15 take K = {0..7,16..23}+kk, lanes 16-31 shifted by +8)
// ---------------------------------------------------------------------------
__device__ __forceinline__ v16bf load_frag16(const __bf16* __restrict__ p) {
    union { v16bf v; uint4 u[2]; } r;
    r.u[0] = *(const uint4*)(p);
    r.u[1] = *(const uint4*)(p + 16);
    return r.v;
}

// ---------------------------------------------------------------------------
// One-time conversion / repack kernels
// ---------------------------------------------------------------------------
__global__ void conv_bf16_kernel(const float* __restrict__ in, __bf16* __restrict__ out, int n) {
    for (int i = blockIdx.x * blockDim.x + threadIdx.x; i < n; i += gridDim.x * blockDim.x)
        out[i] = (__bf16)in[i];
}

// Wa [D][T][TN] -> WaT [T][TN][D] bf16
__global__ void conv_WaT_kernel(const float* __restrict__ Wa, __bf16* __restrict__ WaT) {
    int idx = blockIdx.x * blockDim.x + threadIdx.x;
    if (idx >= TT * TNN * DD) return;
    int t = idx / (TNN * DD);
    int r = idx % (TNN * DD);
    int n = r / DD;
    int k = r % DD;
    WaT[idx] = (__bf16)Wa[((size_t)k * TT + t) * TNN + n];
}

// W [257][2048] -> WT [2048][288] bf16 (zero padded K)
__global__ void conv_WT_kernel(const float* __restrict__ W, __bf16* __restrict__ WT) {
    int idx = blockIdx.x * blockDim.x + threadIdx.x;
    if (idx >= G4 * KWP) return;
    int g = idx / KWP;
    int j = idx % KWP;
    WT[idx] = (j < DD + 1) ? (__bf16)W[(size_t)j * G4 + g] : (__bf16)0.0f;
}

// U [512][2048] -> UT [2048][512] bf16
__global__ void conv_UT_kernel(const float* __restrict__ U, __bf16* __restrict__ UT) {
    int idx = blockIdx.x * blockDim.x + threadIdx.x;
    if (idx >= G4 * HSZ) return;
    int g = idx / HSZ;
    int k = idx % HSZ;
    UT[idx] = (__bf16)U[(size_t)k * G4 + g];
}

__global__ void init_state_kernel(const float* __restrict__ target_init,
                                  float* __restrict__ c, float* __restrict__ h,
                                  __bf16* __restrict__ hbf, float* __restrict__ y_st) {
    int i = blockIdx.x * blockDim.x + threadIdx.x;
    if (i < BB * HSZ) { c[i] = 0.0f; h[i] = 0.0f; hbf[i] = (__bf16)0.0f; }
    if (i < BB) y_st[i] = target_init[i];
}

// ---------------------------------------------------------------------------
// Per-step kernels
// ---------------------------------------------------------------------------

// cUa[b][n] = sum_k c[b][k] * Ua[k][t][n]
__global__ __launch_bounds__(128) void cua_kernel(const float* __restrict__ c,
                                                  const float* __restrict__ Ua,
                                                  int t, float* __restrict__ cUa) {
    int b = blockIdx.x, n = threadIdx.x;
    float acc = 0.0f;
    for (int k = 0; k < HSZ; ++k)
        acc += c[b * HSZ + k] * Ua[((size_t)k * TT + t) * TNN + n];
    cUa[b * TNN + n] = acc;
}

// logits[b][s] = scale * sum_n tanh( (H@wa_t)[b,s,n] + cUa[b,n] + ba_t[n] ) * va[n]
// 256 threads = 8 waves; one wave per 16-row s tile of batch b.
// B (wa_t, [TN][D] = 64KB bf16) is staged once per block into LDS via the
// CDNA5 async-to-LDS path; per N-tile, all 16 ds_load_b128 are issued before
// the 8 back-to-back WMMAs so DScnt waits overlap loads with matrix issue.
__global__ __launch_bounds__(256) void attn_score_kernel(
    const __bf16* __restrict__ Hbf,      // [B][S][D]
    const __bf16* __restrict__ WaT_t,    // [TN][D] for this step
    const float*  __restrict__ cUa,      // [B][TN]
    const float*  __restrict__ ba_t,     // [TN]
    const float*  __restrict__ Va,       // [TN][T]
    int t, float* __restrict__ logits)   // [B][S]
{
    __shared__ __bf16 Bsh[TNN * DD];     // 64 KB

    const int b     = blockIdx.y;
    const int wave  = threadIdx.x >> 5;
    const int lane  = threadIdx.x & 31;
    const int s0    = (blockIdx.x * 8 + wave) * 16;
    const int row   = lane & 15;
    const int khalf = lane >> 4;
    const float scale = 0.04419417382415922f; // 1/sqrt(512)

    // ---- async stage B tile (64KB) into LDS: 256 lanes x 16 iters x 16B ----
    {
        unsigned lbase = (unsigned)(uintptr_t)(&Bsh[0]);
        const char* gbase = (const char*)WaT_t;
#pragma unroll
        for (int it = 0; it < 16; ++it) {
            unsigned byte_off = (threadIdx.x + it * 256) * 16;
            unsigned loff = lbase + byte_off;
            const char* g = gbase + byte_off;
            asm volatile("global_load_async_to_lds_b128 %0, %1, off"
                         :: "v"(loff), "v"(g) : "memory");
        }
        asm volatile("s_wait_asynccnt 0x0" ::: "memory");
    }

    // ---- hoist A fragments (16 rows x 256 K) into registers ----
    const __bf16* Arow = Hbf + ((size_t)b * SS + (s0 + row)) * DD + khalf * 8;
    v16bf afrag[8];
#pragma unroll
    for (int kc = 0; kc < 8; ++kc)
        afrag[kc] = load_frag16(Arow + kc * 32);

    __syncthreads();   // B tile visible to all waves

    float acc[8];
#pragma unroll
    for (int r = 0; r < 8; ++r) acc[r] = 0.0f;

    for (int nt = 0; nt < 8; ++nt) {
        const int n0 = nt * 16;
        const int bbase = (n0 + row) * DD + khalf * 8;

        // issue all LDS loads for this N-tile first (DScnt pipelining)
        union { v16bf v; uint4 u[2]; } bf[8];
#pragma unroll
        for (int kc = 0; kc < 8; ++kc) {
            bf[kc].u[0] = *(const uint4*)&Bsh[bbase + kc * 32];
            bf[kc].u[1] = *(const uint4*)&Bsh[bbase + kc * 32 + 16];
        }
        v8f cacc = {};
#pragma unroll
        for (int kc = 0; kc < 8; ++kc)
            cacc = __builtin_amdgcn_wmma_f32_16x16x32_bf16(
                false, afrag[kc], false, bf[kc].v, (short)0, cacc, false, false);

        const int n    = n0 + row;                 // C-layout: this lane's N column
        const float ad = cUa[b * TNN + n] + ba_t[n];
        const float vn = Va[n * TT + t];
#pragma unroll
        for (int r = 0; r < 8; ++r)
            acc[r] += fast_tanh(cacc[r] + ad) * vn;
    }
    // reduce over the 16 N-lanes of each half-wave
#pragma unroll
    for (int r = 0; r < 8; ++r) {
        float v = acc[r];
        for (int off = 8; off > 0; off >>= 1) v += __shfl_xor(v, off, 32);
        if (row == 0)
            logits[(size_t)b * SS + s0 + r + 8 * khalf] = v * scale;
    }
}

// softmax over S=512 per batch row
__global__ __launch_bounds__(256) void softmax_kernel(const float* __restrict__ logits,
                                                      float* __restrict__ beta) {
    __shared__ float red[8];
    int b = blockIdx.x, tid = threadIdx.x;
    float x0 = logits[b * SS + tid], x1 = logits[b * SS + 256 + tid];
    float m = fmaxf(x0, x1);
    for (int off = 16; off > 0; off >>= 1) m = fmaxf(m, __shfl_xor(m, off, 32));
    if ((tid & 31) == 0) red[tid >> 5] = m;
    __syncthreads();
    float mm = red[0];
#pragma unroll
    for (int i = 1; i < 8; ++i) mm = fmaxf(mm, red[i]);
    __syncthreads();
    float e0 = __expf(x0 - mm), e1 = __expf(x1 - mm);
    float s = e0 + e1;
    for (int off = 16; off > 0; off >>= 1) s += __shfl_xor(s, off, 32);
    if ((tid & 31) == 0) red[tid >> 5] = s;
    __syncthreads();
    float ss = 0.0f;
#pragma unroll
    for (int i = 0; i < 8; ++i) ss += red[i];
    float inv = 1.0f / ss;
    beta[b * SS + tid] = e0 * inv;
    beta[b * SS + 256 + tid] = e1 * inv;
}

// hctx[b][d] = sum_s beta[b][s] * H[b][s][d]   (H is L2-resident)
__global__ __launch_bounds__(256) void context_kernel(const float* __restrict__ beta,
                                                      const float* __restrict__ H,
                                                      float* __restrict__ hctx) {
    int b = blockIdx.x, d = threadIdx.x;
    const float* Hb = H + (size_t)b * SS * DD;
    float acc = 0.0f;
    for (int s = 0; s < SS; ++s)
        acc += beta[b * SS + s] * Hb[(size_t)s * DD + d];
    hctx[b * DD + d] = acc;
}

// feature attention: a[j] = sum_n tanh(hcat[j]*wsa[j][n] + y*usa[n] + bsa[n]) * vsa[n]
// softmax over j in [0,257); output alpha*hcat as bf16 padded to 288 cols.
__global__ __launch_bounds__(256) void feat_attn_kernel(
    const float* __restrict__ hctx, const float* __restrict__ y_prev,
    const float* __restrict__ Wa_sa, const float* __restrict__ Ua_sa,
    const float* __restrict__ ba_sa, const float* __restrict__ Va_sa,
    int t, __bf16* __restrict__ h_att_bf) {
    __shared__ float a_sh[DD + 1];
    __shared__ float h_sh[DD + 1];
    __shared__ float wred[8];
    int b = blockIdx.x, tid = threadIdx.x;
    float yp = y_prev[b];
    const float* urow = Ua_sa + t * ANN;
    const float* brow = ba_sa + t * ANN;
    for (int j = tid; j < DD + 1; j += 256) {
        float hj = (j < DD) ? hctx[b * DD + j] : yp;
        h_sh[j] = hj;
        const float* wrow = Wa_sa + ((size_t)j * TT + t) * ANN;
        float s = 0.0f;
        for (int n = 0; n < ANN; ++n)
            s += fast_tanh(hj * wrow[n] + yp * urow[n] + brow[n]) * Va_sa[n * TT + t];
        a_sh[j] = s;
    }
    __syncthreads();
    float m = -1e30f;
    for (int j = tid; j < DD + 1; j += 256) m = fmaxf(m, a_sh[j]);
    for (int off = 16; off > 0; off >>= 1) m = fmaxf(m, __shfl_xor(m, off, 32));
    if ((tid & 31) == 0) wred[tid >> 5] = m;
    __syncthreads();
    float mm = wred[0];
#pragma unroll
    for (int i = 1; i < 8; ++i) mm = fmaxf(mm, wred[i]);
    __syncthreads();
    float s = 0.0f;
    for (int j = tid; j < DD + 1; j += 256) s += __expf(a_sh[j] - mm);
    for (int off = 16; off > 0; off >>= 1) s += __shfl_xor(s, off, 32);
    if ((tid & 31) == 0) wred[tid >> 5] = s;
    __syncthreads();
    float ss = 0.0f;
#pragma unroll
    for (int i = 0; i < 8; ++i) ss += wred[i];
    float inv = 1.0f / ss;
    for (int j = tid; j < KWP; j += 256) {
        float v = 0.0f;
        if (j < DD + 1) v = __expf(a_sh[j] - mm) * inv * h_sh[j];
        h_att_bf[(size_t)b * KWP + j] = (__bf16)v;
    }
}

// gates[b][g] = bias[g] + h_att@W + h_lstm@U   (two bf16 WMMA K-loops, f32 accum)
__global__ __launch_bounds__(128) void gates_kernel(
    const __bf16* __restrict__ Abf,   // [B][288]
    const __bf16* __restrict__ WTb,   // [2048][288]
    const __bf16* __restrict__ Hlbf,  // [B][512]
    const __bf16* __restrict__ UTb,   // [2048][512]
    const float*  __restrict__ bias,
    float* __restrict__ gates) {
    const int wave  = threadIdx.x >> 5;
    const int lane  = threadIdx.x & 31;
    const int n0    = (blockIdx.x * 4 + wave) * 16;
    const int m0    = blockIdx.y * 16;
    const int row   = lane & 15;
    const int khalf = lane >> 4;

    v8f cacc = {};
    {
        const __bf16* Ar = Abf + (size_t)(m0 + row) * KWP + khalf * 8;
        const __bf16* Br = WTb + (size_t)(n0 + row) * KWP + khalf * 8;
#pragma unroll
        for (int kk = 0; kk < KWP; kk += 32) {
            v16bf a = load_frag16(Ar + kk);
            v16bf bm = load_frag16(Br + kk);
            cacc = __builtin_amdgcn_wmma_f32_16x16x32_bf16(
                false, a, false, bm, (short)0, cacc, false, false);
        }
    }
    {
        const __bf16* Ar = Hlbf + (size_t)(m0 + row) * HSZ + khalf * 8;
        const __bf16* Br = UTb + (size_t)(n0 + row) * HSZ + khalf * 8;
#pragma unroll
        for (int kk = 0; kk < HSZ; kk += 32) {
            v16bf a = load_frag16(Ar + kk);
            v16bf bm = load_frag16(Br + kk);
            cacc = __builtin_amdgcn_wmma_f32_16x16x32_bf16(
                false, a, false, bm, (short)0, cacc, false, false);
        }
    }
    const int n = n0 + row;
    const float bn = bias[n];
#pragma unroll
    for (int r = 0; r < 8; ++r) {
        int m = m0 + r + 8 * khalf;
        gates[(size_t)m * G4 + n] = cacc[r] + bn;
    }
}

// LSTM pointwise: c/h update, bf16 mirror of h, hidden_seq output
__global__ __launch_bounds__(256) void lstm_cell_kernel(
    const float* __restrict__ gates, float* __restrict__ c, float* __restrict__ h,
    __bf16* __restrict__ hbf, float* __restrict__ hidden_out, int t) {
    int b = blockIdx.x;
    const float* g = gates + (size_t)b * G4;
    for (int k = threadIdx.x; k < HSZ; k += 256) {
        float i_t = fast_sigmoid(g[k]);
        float f_t = fast_sigmoid(g[HSZ + k]);
        float g_t = fast_tanh(g[2 * HSZ + k]);
        float o_t = fast_sigmoid(g[3 * HSZ + k]);
        float cn = f_t * c[b * HSZ + k] + i_t * g_t;
        float hn = o_t * fast_tanh(cn);
        c[b * HSZ + k] = cn;
        h[b * HSZ + k] = hn;
        hbf[b * HSZ + k] = (__bf16)hn;
        hidden_out[((size_t)b * TT + t) * HSZ + k] = hn;
    }
}

// y = h@fc_w.T + fc_b; update y state and y output
__global__ __launch_bounds__(256) void y_kernel(
    const float* __restrict__ h, const float* __restrict__ fc_w,
    const float* __restrict__ fc_b, float* __restrict__ y_st,
    float* __restrict__ y_out, int t) {
    __shared__ float wred[8];
    int b = blockIdx.x, tid = threadIdx.x;
    float s = h[b * HSZ + tid] * fc_w[tid] + h[b * HSZ + 256 + tid] * fc_w[256 + tid];
    for (int off = 16; off > 0; off >>= 1) s += __shfl_xor(s, off, 32);
    if ((tid & 31) == 0) wred[tid >> 5] = s;
    __syncthreads();
    if (tid == 0) {
        float ss = 0.0f;
#pragma unroll
        for (int i = 0; i < 8; ++i) ss += wred[i];
        ss += fc_b[0];
        y_st[b] = ss;
        y_out[b * TT + t] = ss;
    }
}

// ---------------------------------------------------------------------------
extern "C" void kernel_launch(void* const* d_in, const int* in_sizes, int n_in,
                              void* d_out, int out_size, void* d_ws, size_t ws_size,
                              hipStream_t stream) {
    const float* H        = (const float*)d_in[0];
    const float* tgt0     = (const float*)d_in[1];
    const float* Wa       = (const float*)d_in[2];
    const float* Ua       = (const float*)d_in[3];
    const float* ba       = (const float*)d_in[4];
    const float* Va       = (const float*)d_in[5];
    const float* Wa_sa    = (const float*)d_in[6];
    const float* Ua_sa    = (const float*)d_in[7];
    const float* ba_sa    = (const float*)d_in[8];
    const float* Va_sa    = (const float*)d_in[9];
    const float* W        = (const float*)d_in[10];
    const float* U        = (const float*)d_in[11];
    const float* bias     = (const float*)d_in[12];
    const float* fc_w     = (const float*)d_in[13];
    const float* fc_b     = (const float*)d_in[14];

    float* y_out   = (float*)d_out;             // [B][T]
    float* hid_out = y_out + (size_t)BB * TT;   // [B][T][HS]

    size_t off = 0;
    auto alloc = [&](size_t bytes) -> void* {
        off = (off + 255) & ~(size_t)255;
        void* p = (char*)d_ws + off;
        off += bytes;
        return p;
    };
    __bf16* Hbf      = (__bf16*)alloc((size_t)BB * SS * DD * 2);
    __bf16* WaT      = (__bf16*)alloc((size_t)TT * TNN * DD * 2);
    __bf16* WT       = (__bf16*)alloc((size_t)G4 * KWP * 2);
    __bf16* UT       = (__bf16*)alloc((size_t)G4 * HSZ * 2);
    float*  c_st     = (float*)alloc((size_t)BB * HSZ * 4);
    float*  h_st     = (float*)alloc((size_t)BB * HSZ * 4);
    __bf16* hbf      = (__bf16*)alloc((size_t)BB * HSZ * 2);
    float*  y_st     = (float*)alloc((size_t)BB * 4);
    float*  cUa      = (float*)alloc((size_t)BB * TNN * 4);
    float*  logits   = (float*)alloc((size_t)BB * SS * 4);
    float*  beta     = (float*)alloc((size_t)BB * SS * 4);
    float*  hctx     = (float*)alloc((size_t)BB * DD * 4);
    __bf16* h_att_bf = (__bf16*)alloc((size_t)BB * KWP * 2);
    float*  gates    = (float*)alloc((size_t)BB * G4 * 4);
    (void)ws_size; (void)in_sizes; (void)n_in; (void)out_size;

    // one-time repacks
    conv_bf16_kernel<<<4096, 256, 0, stream>>>(H, Hbf, BB * SS * DD);
    conv_WaT_kernel<<<(TT * TNN * DD + 255) / 256, 256, 0, stream>>>(Wa, WaT);
    conv_WT_kernel<<<(G4 * KWP + 255) / 256, 256, 0, stream>>>(W, WT);
    conv_UT_kernel<<<(G4 * HSZ + 255) / 256, 256, 0, stream>>>(U, UT);
    init_state_kernel<<<(BB * HSZ + 255) / 256, 256, 0, stream>>>(tgt0, c_st, h_st, hbf, y_st);

    for (int t = 0; t < TT; ++t) {
        cua_kernel<<<BB, TNN, 0, stream>>>(c_st, Ua, t, cUa);
        attn_score_kernel<<<dim3(SS / 128, BB), 256, 0, stream>>>(
            Hbf, WaT + (size_t)t * TNN * DD, cUa, ba + t * TNN, Va, t, logits);
        softmax_kernel<<<BB, 256, 0, stream>>>(logits, beta);
        context_kernel<<<BB, DD, 0, stream>>>(beta, H, hctx);
        feat_attn_kernel<<<BB, 256, 0, stream>>>(hctx, y_st, Wa_sa, Ua_sa, ba_sa, Va_sa, t, h_att_bf);
        gates_kernel<<<dim3(G4 / 64, BB / 16), 128, 0, stream>>>(h_att_bf, WT, hbf, UT, bias, gates);
        lstm_cell_kernel<<<BB, 256, 0, stream>>>(gates, c_st, h_st, hbf, hid_out, t);
        y_kernel<<<BB, 256, 0, stream>>>(h_st, fc_w, fc_b, y_st, y_out, t);
    }
}
